// HyperGraphMessageNet_5892695130345
// MI455X (gfx1250) — matrix operand order repaired and checked by reading, not compiled, not run on hardware
//
#include <hip/hip_runtime.h>
#include <hip/hip_bf16.h>
#include <math.h>

#define E_N 16384
#define S_N 8192
#define H_N 128
#define L_N 4

typedef unsigned short u16;
typedef __attribute__((ext_vector_type(16))) __bf16 bf16x16;
typedef __attribute__((ext_vector_type(8)))  float  f32x8;

union FragBF { int4 q[2]; bf16x16 v; };

__device__ __forceinline__ u16 f2bf(float f) {
  union { float f; unsigned u; } x; x.f = f;
  unsigned r = x.u + 0x7FFFu + ((x.u >> 16) & 1u);
  return (u16)(r >> 16);
}

// ---------------------------------------------------------------- utilities
__global__ void k_zero(float* p, int n) {
  int i = blockIdx.x * 256 + threadIdx.x;
  if (i < n) p[i] = 0.f;
}

// src [rows][H] f32 -> dstF (f32), dstB (bf16 row-major), dstT (bf16 [H][rows])
__global__ void k_quant_copy(const float* __restrict__ src, float* __restrict__ dstF,
                             u16* __restrict__ dstB, u16* __restrict__ dstT,
                             int rows) {
  int i = blockIdx.x * 256 + threadIdx.x;
  int n = rows * H_N;
  if (i >= n) return;
  int m = i / H_N, c = i % H_N;
  float v = src[i];
  dstF[i] = v;
  u16 b = f2bf(v);
  dstB[i] = b;
  dstT[(size_t)c * rows + m] = b;
}

// w [L][2H][H] f32  ->  wT [L][H][2H] bf16  (N-major, K contiguous)
__global__ void k_wT(const float* __restrict__ w, u16* __restrict__ wT, int n) {
  int i = blockIdx.x * 256 + threadIdx.x;
  if (i >= n) return;
  int l = i / (2 * H_N * H_N);
  int r = i % (2 * H_N * H_N);
  int k = r / H_N;
  int c = r % H_N;
  wT[(size_t)l * (2 * H_N * H_N) + (size_t)c * (2 * H_N) + k] = f2bf(w[i]);
}

// inc [E][S] f32 -> incT [S][E] bf16, incR [E][S] bf16, + degree vectors
__global__ void k_prep_inc(const float* __restrict__ inc, u16* __restrict__ incT,
                           u16* __restrict__ incR,
                           float* __restrict__ node_deg, float* __restrict__ edge_deg) {
  __shared__ float tile[64][65];
  const int t  = threadIdx.x;
  const int s0 = blockIdx.x * 64, e0 = blockIdx.y * 64;
  const int sl = t & 63, eq = t >> 6;
  #pragma unroll
  for (int r = 0; r < 16; ++r) {
    int el = r * 4 + eq;
    tile[sl][el] = inc[(size_t)(e0 + el) * S_N + s0 + sl];
  }
  __syncthreads();
  if (t < 64) {                       // node_deg[e] = sum_s inc[e][s]
    float s = 0.f;
    #pragma unroll 8
    for (int j = 0; j < 64; ++j) s += tile[j][t];
    atomicAdd(&node_deg[e0 + t], s);
  } else if (t < 128) {               // edge_deg[s] = sum_e inc[e][s]
    int sr = t - 64; float s = 0.f;
    #pragma unroll 8
    for (int j = 0; j < 64; ++j) s += tile[sr][j];
    atomicAdd(&edge_deg[s0 + sr], s);
  }
  const int ro = t >> 2, g = t & 3;
  {   // incT[s][e]
    __align__(16) u16 buf[16];
    #pragma unroll
    for (int j = 0; j < 16; ++j) buf[j] = f2bf(tile[ro][g * 16 + j]);
    u16* dst = incT + (size_t)(s0 + ro) * E_N + e0 + g * 16;
    *(int4*)(dst)     = *(const int4*)&buf[0];
    *(int4*)(dst + 8) = *(const int4*)&buf[8];
  }
  {   // incR[e][s]
    __align__(16) u16 buf[16];
    #pragma unroll
    for (int j = 0; j < 16; ++j) buf[j] = f2bf(tile[g * 16 + j][ro]);
    u16* dst = incR + (size_t)(e0 + ro) * S_N + s0 + g * 16;
    *(int4*)(dst)     = *(const int4*)&buf[0];
    *(int4*)(dst + 8) = *(const int4*)&buf[8];
  }
}

// ------------------------------------------------- message GEMM (bf16 WMMA)
// parts[split][M][128] = A(MxK) @ B(Kx128)
// A: natural row-major [M][K].  BT: transposed emb [128][K].
// K split over gridDim.y.  Ping-pong LDS (compile-time buffer index via 2x
// unroll -> all ds addresses are base + immediate), register double buffering,
// branchless one-step-ahead global loads, 1 barrier per K-step.

#define LA_SZ 2560                     // 64*40 u16 per buffer
#define LB_SZ 5120                     // 128*40 u16 per buffer

__device__ __forceinline__ f32x8 wmma_bf16(const FragBF& a, const FragBF& b,
                                           f32x8 c) {
  return __builtin_amdgcn_wmma_f32_16x16x32_bf16(
      false, a.v, false, b.v, (short)0, c, false, false);
}

template <int P>
__device__ __forceinline__ void gemm_stage(u16* aDst, u16* bDst0, u16* bDst1,
                                           int4 a, int4 b0, int4 b1) {
  *(int4*)(aDst  + P * LA_SZ) = a;
  *(int4*)(bDst0 + P * LB_SZ) = b0;
  *(int4*)(bDst1 + P * LB_SZ) = b1;
}

template <int P>
__device__ __forceinline__ void gemm_step(const u16* apB, const u16* bpB,
                                          f32x8 (&acc)[4]) {
  FragBF a, b0, b1, b2, b3;
  a.q[0]  = *(const int4*)(apB + P * LA_SZ);
  a.q[1]  = *(const int4*)(apB + P * LA_SZ + 8);
  b0.q[0] = *(const int4*)(bpB + P * LB_SZ);
  b0.q[1] = *(const int4*)(bpB + P * LB_SZ + 8);
  b1.q[0] = *(const int4*)(bpB + P * LB_SZ + 640);
  b1.q[1] = *(const int4*)(bpB + P * LB_SZ + 648);
  b2.q[0] = *(const int4*)(bpB + P * LB_SZ + 1280);
  b2.q[1] = *(const int4*)(bpB + P * LB_SZ + 1288);
  b3.q[0] = *(const int4*)(bpB + P * LB_SZ + 1920);
  b3.q[1] = *(const int4*)(bpB + P * LB_SZ + 1928);
  acc[0] = wmma_bf16(a, b0, acc[0]);
  acc[1] = wmma_bf16(a, b1, acc[1]);
  acc[2] = wmma_bf16(a, b2, acc[2]);
  acc[3] = wmma_bf16(a, b3, acc[3]);
}

__global__ void k_msg_gemm(const u16* __restrict__ Arow, const u16* __restrict__ BT,
                           float* __restrict__ parts, int M, int K) {
  __shared__ __align__(16) u16 lds[2 * LA_SZ + 2 * LB_SZ];   // 30720B
  u16* lA = lds;                       // [2][64][40]
  u16* lB = lds + 2 * LA_SZ;           // [2][128][40]
  const int t    = threadIdx.x;
  const int m0   = blockIdx.x * 64;
  const int Kc   = K / gridDim.y;
  const int kBeg = Kc * blockIdx.y;
  const int wid  = t >> 5, lane = t & 31;        // wave32
  const int wm   = wid & 3, wn = wid >> 2;
  const int ln   = lane & 15, hi = lane >> 4;

  // staging source pointers (coalesced b128)
  const int am  = t >> 2, akg = t & 3;
  const u16* aSrc = Arow + (size_t)(m0 + am) * K + kBeg + akg * 8;
  const int bi0 = t,        bn0 = bi0 >> 2, bkg0 = bi0 & 3;
  const int bi1 = t + 256,  bn1 = bi1 >> 2, bkg1 = bi1 & 3;
  const u16* bSrc0 = BT + (size_t)bn0 * K + kBeg + bkg0 * 8;
  const u16* bSrc1 = BT + (size_t)bn1 * K + kBeg + bkg1 * 8;

  // LDS staging destinations / fragment bases (loop-invariant)
  u16* aDst  = lA + am * 40 + akg * 8;
  u16* bDst0 = lB + bn0 * 40 + bkg0 * 8;
  u16* bDst1 = lB + bn1 * 40 + bkg1 * 8;
  const u16* apB = lA + (wm * 16 + ln) * 40 + hi * 16;
  const u16* bpB = lB + (wn * 64 + ln) * 40 + hi * 16;

  f32x8 acc[4];
  #pragma unroll
  for (int tt = 0; tt < 4; ++tt)
    #pragma unroll
    for (int r = 0; r < 8; ++r) acc[tt][r] = 0.f;

  int4 aC  = *(const int4*)aSrc;
  int4 b0C = *(const int4*)bSrc0;
  int4 b1C = *(const int4*)bSrc1;

  const int nIter = Kc >> 5;           // 64 (edge) or 32 (node): even
  int off = 32;
  for (int it = 0; it < nIter; it += 2) {
    // ---- step P=0
    int4 aN  = *(const int4*)(aSrc + off);      // one step ahead (branchless;
    int4 b0N = *(const int4*)(bSrc0 + off);     //  final over-read stays in ws)
    int4 b1N = *(const int4*)(bSrc1 + off);
    gemm_stage<0>(aDst, bDst0, bDst1, aC, b0C, b1C);
    __syncthreads();
    gemm_step<0>(apB, bpB, acc);
    off += 32;
    // ---- step P=1
    aC  = *(const int4*)(aSrc + off);
    b0C = *(const int4*)(bSrc0 + off);
    b1C = *(const int4*)(bSrc1 + off);
    gemm_stage<1>(aDst, bDst0, bDst1, aN, b0N, b1N);
    __syncthreads();
    gemm_step<1>(apB, bpB, acc);
    off += 32;
  }

  float* out = parts + (size_t)blockIdx.y * (size_t)M * H_N;
  #pragma unroll
  for (int tt = 0; tt < 4; ++tt) {
    int col = wn * 64 + tt * 16 + ln;
    #pragma unroll
    for (int r = 0; r < 8; ++r) {
      int row = m0 + wm * 16 + hi * 8 + r;
      out[(size_t)row * H_N + col] = acc[tt][r];
    }
  }
}

// ---------------- fused: split-reduce msg, /deg, concat MLP (WMMA), bias,
// ReLU, LayerNorm, residual; writes f32 + bf16 (row & transposed) embeddings.
__global__ void k_mlp_ln(const float* __restrict__ parts, int nparts, int M,
                         const float* __restrict__ deg,
                         float* __restrict__ embF, u16* __restrict__ embB,
                         u16* __restrict__ embT,
                         const u16* __restrict__ WT,   // [128][256] bf16 (layer)
                         const float* __restrict__ bias,
                         const float* __restrict__ gamma,
                         const float* __restrict__ beta) {
  __shared__ __align__(16) unsigned char smem[64 * 264 * 2];  // 33792B dual use
  __shared__ float stat[128];
  u16*   lA = (u16*)smem;    // [64][264] bf16 concat input
  float* lH = (float*)smem;  // [64][132] f32 hidden / output staging
  const int t  = threadIdx.x;
  const int m0 = blockIdx.x * 64;

  #pragma unroll
  for (int i = 0; i < 4; ++i) {                 // emb rows -> cols 0..127
    int idx = t + 256 * i;
    int m = idx >> 4, ng = idx & 15;
    *(int4*)&lA[m * 264 + ng * 8] =
        *(const int4*)(embB + (size_t)(m0 + m) * H_N + ng * 8);
  }
  #pragma unroll
  for (int i = 0; i < 8; ++i) {                 // msg sum/deg -> cols 128..255
    int idx = t + 256 * i;                      // 0..2047
    int m = idx >> 5, cq = idx & 31;
    float4 s = make_float4(0.f, 0.f, 0.f, 0.f);
    for (int p = 0; p < nparts; ++p) {
      float4 v = *(const float4*)(parts + (size_t)p * M * H_N +
                                  (size_t)(m0 + m) * H_N + cq * 4);
      s.x += v.x; s.y += v.y; s.z += v.z; s.w += v.w;
    }
    float d = deg[m0 + m]; d = d > 1e-6f ? d : 1e-6f;
    float inv = 1.f / d;
    u16* dst = &lA[m * 264 + 128 + cq * 4];
    dst[0] = f2bf(s.x * inv); dst[1] = f2bf(s.y * inv);
    dst[2] = f2bf(s.z * inv); dst[3] = f2bf(s.w * inv);
  }
  __syncthreads();

  const int wid = t >> 5, lane = t & 31;
  const int wm = wid & 3, wn = wid >> 2;
  const int ln = lane & 15, hi = lane >> 4;
  f32x8 acc[4];
  #pragma unroll
  for (int tt = 0; tt < 4; ++tt)
    #pragma unroll
    for (int r = 0; r < 8; ++r) acc[tt][r] = 0.f;

  #pragma unroll
  for (int ks = 0; ks < 8; ++ks) {
    int k0 = ks * 32;
    FragBF a;
    const u16* ap = &lA[(wm * 16 + ln) * 264 + k0 + hi * 16];
    a.q[0] = *(const int4*)ap;
    a.q[1] = *(const int4*)(ap + 8);
    #pragma unroll
    for (int tt = 0; tt < 4; ++tt) {
      int n = wn * 64 + tt * 16 + ln;
      const u16* bp = WT + (size_t)n * 256 + k0 + hi * 16;  // L2-hot
      FragBF b;
      b.q[0] = *(const int4*)bp;
      b.q[1] = *(const int4*)(bp + 8);
      acc[tt] = wmma_bf16(a, b, acc[tt]);
    }
  }
  __syncthreads();  // all lA reads done before overwriting with lH

  #pragma unroll
  for (int tt = 0; tt < 4; ++tt) {
    int col = wn * 64 + tt * 16 + ln;
    float bv = bias[col];
    #pragma unroll
    for (int r = 0; r < 8; ++r) {
      int row = wm * 16 + hi * 8 + r;
      float h = acc[tt][r] + bv;
      lH[row * 132 + col] = h > 0.f ? h : 0.f;
    }
  }
  __syncthreads();
  if (t < 64) {
    float s = 0.f, ss = 0.f;
    #pragma unroll 8
    for (int c = 0; c < 128; ++c) { float h = lH[t * 132 + c]; s += h; ss += h * h; }
    float mu  = s * (1.f / 128.f);
    float var = ss * (1.f / 128.f) - mu * mu;
    stat[t]      = mu;
    stat[64 + t] = rsqrtf(var + 1e-5f);
  }
  __syncthreads();
  #pragma unroll
  for (int i = 0; i < 32; ++i) {       // residual + LN, write f32 + bf16(row)
    int idx = t + 256 * i;
    int m = idx >> 7, c = idx & 127;
    float h = lH[m * 132 + c];
    size_t gi = (size_t)(m0 + m) * H_N + c;
    float v = embF[gi] + (h - stat[m]) * stat[64 + m] * gamma[c] + beta[c];
    embF[gi] = v;
    embB[gi] = f2bf(v);
    lH[m * 132 + c] = v;               // stage for transposed write
  }
  __syncthreads();
  #pragma unroll
  for (int i = 0; i < 32; ++i) {       // bf16 transposed copy [H][M]
    int idx = t + 256 * i;
    int c = idx >> 6, m = idx & 63;
    embT[(size_t)c * M + m0 + m] = f2bf(lH[m * 132 + c]);
  }
}

__global__ void k_decode(const float* __restrict__ embF, const float* __restrict__ dw,
                         const float* __restrict__ db, float* __restrict__ out) {
  int s = blockIdx.x * 256 + threadIdx.x;
  if (s >= S_N) return;
  const float4* r = (const float4*)(embF + (size_t)s * H_N);
  const float4* w = (const float4*)dw;
  float acc = 0.f;
  #pragma unroll
  for (int i = 0; i < 32; ++i) {
    float4 a = r[i], b = w[i];
    acc += a.x * b.x + a.y * b.y + a.z * b.z + a.w * b.w;
  }
  float logit = acc + db[0];
  out[s] = 1.f / (1.f + expf(-0.7f * logit));
}

// ---------------------------------------------------------------- launch
extern "C" void kernel_launch(void* const* d_in, const int* in_sizes, int n_in,
                              void* d_out, int out_size, void* d_ws, size_t ws_size,
                              hipStream_t stream) {
  const float* inc     = (const float*)d_in[0];
  const float* nemb0   = (const float*)d_in[1];
  const float* eemb0   = (const float*)d_in[2];
  const float* w_edge  = (const float*)d_in[3];
  const float* b_edge  = (const float*)d_in[4];
  const float* g_edge  = (const float*)d_in[5];
  const float* bb_edge = (const float*)d_in[6];
  const float* w_node  = (const float*)d_in[7];
  const float* b_node  = (const float*)d_in[8];
  const float* g_node  = (const float*)d_in[9];
  const float* bb_node = (const float*)d_in[10];
  const float* dec_w   = (const float*)d_in[11];
  const float* dec_b   = (const float*)d_in[12];
  (void)in_sizes; (void)n_in; (void)out_size; (void)ws_size;

  char* ws = (char*)d_ws;
  size_t off = 0;
  auto alloc = [&](size_t bytes) -> void* {
    void* p = ws + off;
    off = (off + bytes + 255) & ~(size_t)255;
    return p;
  };
  u16*   incT  = (u16*)alloc((size_t)S_N * E_N * 2);   // [S][E]
  u16*   incR  = (u16*)alloc((size_t)E_N * S_N * 2);   // [E][S]
  u16*   wTe   = (u16*)alloc((size_t)L_N * 2 * H_N * H_N * 2);
  u16*   wTn   = (u16*)alloc((size_t)L_N * 2 * H_N * H_N * 2);
  float* edeg  = (float*)alloc((size_t)S_N * 4);
  float* ndeg  = (float*)alloc((size_t)E_N * 4);
  float* nF    = (float*)alloc((size_t)E_N * H_N * 4);
  float* eF    = (float*)alloc((size_t)S_N * H_N * 4);
  u16*   nB    = (u16*)alloc((size_t)E_N * H_N * 2);   // row-major bf16
  u16*   eB    = (u16*)alloc((size_t)S_N * H_N * 2);
  u16*   nT    = (u16*)alloc((size_t)H_N * E_N * 2);   // transposed bf16 [H][E]
  u16*   eT    = (u16*)alloc((size_t)H_N * S_N * 2);   // [H][S]
  float* parts = (float*)alloc((size_t)8 * E_N * H_N * 4);

  k_zero<<<(S_N + 255) / 256, 256, 0, stream>>>(edeg, S_N);
  k_zero<<<(E_N + 255) / 256, 256, 0, stream>>>(ndeg, E_N);
  k_prep_inc<<<dim3(S_N / 64, E_N / 64), 256, 0, stream>>>(inc, incT, incR, ndeg, edeg);
  k_quant_copy<<<(E_N * H_N + 255) / 256, 256, 0, stream>>>(nemb0, nF, nB, nT, E_N);
  k_quant_copy<<<(S_N * H_N + 255) / 256, 256, 0, stream>>>(eemb0, eF, eB, eT, S_N);
  int nw = L_N * 2 * H_N * H_N;
  k_wT<<<(nw + 255) / 256, 256, 0, stream>>>(w_edge, wTe, nw);
  k_wT<<<(nw + 255) / 256, 256, 0, stream>>>(w_node, wTn, nw);

  for (int l = 0; l < L_N; ++l) {
    // edge_msg = inc^T @ node_emb : A=incT [S][E], B=nT [128][E]
    k_msg_gemm<<<dim3(S_N / 64, 8), 256, 0, stream>>>(incT, nT, parts, S_N, E_N);
    k_mlp_ln<<<S_N / 64, 256, 0, stream>>>(parts, 8, S_N, edeg, eF, eB, eT,
                                           wTe + (size_t)l * 2 * H_N * H_N,
                                           b_edge + l * H_N, g_edge + l * H_N,
                                           bb_edge + l * H_N);
    // node_msg = inc @ edge_emb : A=incR [E][S], B=eT [128][S]
    k_msg_gemm<<<dim3(E_N / 64, 8), 256, 0, stream>>>(incR, eT, parts, E_N, S_N);
    k_mlp_ln<<<E_N / 64, 256, 0, stream>>>(parts, 8, E_N, ndeg, nF, nB, nT,
                                           wTn + (size_t)l * 2 * H_N * H_N,
                                           b_node + l * H_N, g_node + l * H_N,
                                           bb_node + l * H_N);
  }
  k_decode<<<S_N / 256, 256, 0, stream>>>(eF, dec_w, dec_b, (float*)d_out);
}
